// CoupledFlowMatching_77240691851754
// MI455X (gfx1250) — matrix operand order repaired and checked by reading, not compiled, not run on hardware
//
#include <hip/hip_runtime.h>
#include <hip/hip_bf16.h>

typedef __attribute__((ext_vector_type(16))) _Float16 v16h;
typedef __attribute__((ext_vector_type(8)))  _Float16 v8h;
typedef __attribute__((ext_vector_type(8)))  float    v8f;

#define NSTEPS   32
#define DELTA_F  (1.0f / 32.0f)
#define LOG2PI_F 1.8378770664093453f

__device__ __forceinline__ v8f wmma_f16(v16h a, v16h b, v8f c) {
  // D = A(16x32 f16) * B(32x16 f16) + C(16x16 f32)
  return __builtin_amdgcn_wmma_f32_16x16x32_f16(false, a, false, b, (short)0, c,
                                                false, false);
}

// A operand (16x32 f16): lane holds row M=lane%16; VGPR0-3 => K = g*8+0..7,
// VGPR4-7 => K = 16+g*8+0..7  (g = lane/16).  Two ds_load_b128 per lane.
__device__ __forceinline__ v16h load_a16(const _Float16* row, int k0, int g) {
  const _Float16* p = row + k0 + g * 8;
  v8h lo = *(const v8h*)(p);
  v8h hi = *(const v8h*)(p + 16);
  v16h r;
#pragma unroll
  for (int i = 0; i < 8; ++i) { r[i] = lo[i]; r[i + 8] = hi[i]; }
  return r;
}

// B operand (32x16 f16): lane holds column N=lane%16; K = g*8+0..7, 16+g*8+0..7.
// Buffer is [N][K] row-major.  Two ds_load_b128 per lane.
__device__ __forceinline__ v16h load_b16(const _Float16* row, int k0, int g) {
  const _Float16* p = row + k0 + g * 8;
  v8h lo = *(const v8h*)(p);
  v8h hi = *(const v8h*)(p + 16);
  v16h r;
#pragma unroll
  for (int i = 0; i < 8; ++i) { r[i] = lo[i]; r[i + 8] = hi[i]; }
  return r;
}

__device__ __forceinline__ float red16(float p) {
  p += __shfl_xor(p, 1, 32);
  p += __shfl_xor(p, 2, 32);
  p += __shfl_xor(p, 4, 32);
  p += __shfl_xor(p, 8, 32);
  return p;
}

__global__ void __launch_bounds__(256, 1)
cfm_loglik_kernel(const float* __restrict__ sg, const float* __restrict__ a0g,
                  const int* __restrict__ radg,
                  const float* __restrict__ W1g, const float* __restrict__ b1g,
                  const float* __restrict__ W2g, const float* __restrict__ b2g,
                  const float* __restrict__ W3g, const float* __restrict__ b3g,
                  float* __restrict__ outg, int Btot) {
  extern __shared__ _Float16 smem[];

  // ---- shared weight tiles (f16), [N][K] layouts for WMMA B operands ----
  _Float16* W1T = smem;               // [128][96]  fwd L1: B[k][n]=W1[k][n]
  _Float16* W2T = W1T + 128 * 96;     // [128][128] fwd L2
  _Float16* W3T = W2T + 128 * 128;    // [32][128]  fwd L3
  _Float16* W2R = W3T + 32 * 128;     // [128][128] bwd L2: B = W2^T -> W2 row-major
  _Float16* W3R = W2R + 128 * 128;    // [128][32]  bwd L3: B = W3^T -> W3 row-major
  _Float16* W1A = W3R + 128 * 32;     // [16][128]  bwd L1: rows 0..15 of W1
  float* b1s = (float*)(W1A + 16 * 128);
  float* b2s = b1s + 128;
  float* b3s = b2s + 128;

  const int tid  = threadIdx.x;
  const int wave = tid >> 5;
  const int lane = tid & 31;
  const int l16  = lane & 15;
  const int g    = lane >> 4;

  // ---- per-wave activation staging (f16, row-major [M][K]) ----
  _Float16* pw  = (_Float16*)(b3s + 32) + wave * 9984;
  _Float16* xb  = pw;                 // [16][96]  (a | s | t | pad)
  _Float16* h1b = xb  + 16 * 96;      // [16][128]
  _Float16* h2b = h1b + 16 * 128;     // [16][128]
  _Float16* gb1 = h2b + 16 * 128;     // [16][128]
  _Float16* gb2 = gb1 + 16 * 128;     // [16][128]
  _Float16* ub  = gb2 + 16 * 128;     // [16][16]

  // ---- cooperative weight load + f16 convert ----
  for (int i = tid; i < 128 * 96; i += 256) {
    int n = i / 96, k = i - n * 96;
    W1T[i] = (k < 81) ? (_Float16)W1g[k * 128 + n] : (_Float16)0.0f;
  }
  for (int i = tid; i < 128 * 128; i += 256) {
    int n = i >> 7, k = i & 127;
    W2T[i] = (_Float16)W2g[k * 128 + n];
  }
  for (int i = tid; i < 32 * 128; i += 256) {
    int n = i >> 7, k = i & 127;
    W3T[i] = (_Float16)W3g[k * 32 + n];
  }
  for (int i = tid; i < 128 * 128; i += 256) W2R[i] = (_Float16)W2g[i];
  for (int i = tid; i < 128 * 32;  i += 256) W3R[i] = (_Float16)W3g[i];
  for (int i = tid; i < 16 * 128;  i += 256) W1A[i] = (_Float16)W1g[i];
  if (tid < 128) { b1s[tid] = b1g[tid]; b2s[tid] = b2g[tid]; }
  if (tid < 32)  { b3s[tid] = b3g[tid]; }
  __syncthreads();  // weights visible to all waves; the only block-wide sync

  const int rowbase = blockIdx.x * 128 + wave * 16;

  // stage s into xb cols 16..79 (constant over all steps)
#pragma unroll 4
  for (int j = 0; j < 32; ++j)
    xb[l16 * 96 + 16 + g * 32 + j] =
        (_Float16)sg[(size_t)(rowbase + l16) * 64 + g * 32 + j];
  if (g == 0) {
#pragma unroll
    for (int j = 81; j < 96; ++j) xb[l16 * 96 + j] = (_Float16)0.0f;
  }

  // master fp32 state in C-layout: element (M = r + 8g, dim = l16)
  float a_reg[8], ld_reg[8];
#pragma unroll
  for (int r = 0; r < 8; ++r) {
    a_reg[r] = a0g[(size_t)(rowbase + r + g * 8) * 16 + l16];
    ld_reg[r] = 0.0f;
  }

  // All activation buffers are private to this wave; same-wave LDS ordering
  // is guaranteed by hardware + compiler-inserted s_wait_dscnt, so no
  // barriers are needed inside the step loop.
#pragma unroll 1
  for (int step = 0; step < NSTEPS; ++step) {
    float tmid = ((float)step + 0.5f) * DELTA_F;
#pragma unroll
    for (int r = 0; r < 8; ++r)
      xb[(r + g * 8) * 96 + l16] = (_Float16)a_reg[r];
    if (g == 0) xb[l16 * 96 + 80] = (_Float16)tmid;

    // ================= forward L1: x[16x96] @ W1 -> h1[16x128] ==========
    {
      v16h A0 = load_a16(xb + l16 * 96, 0, g);
      v16h A1 = load_a16(xb + l16 * 96, 32, g);
      v16h A2 = load_a16(xb + l16 * 96, 64, g);
#pragma unroll
      for (int nt = 0; nt < 8; ++nt) {
        const _Float16* brow = W1T + (nt * 16 + l16) * 96;
        v8f C = {};
        C = wmma_f16(A0, load_b16(brow, 0, g), C);
        C = wmma_f16(A1, load_b16(brow, 32, g), C);
        C = wmma_f16(A2, load_b16(brow, 64, g), C);
        float bia = b1s[nt * 16 + l16];
#pragma unroll
        for (int r = 0; r < 8; ++r) {
          float h = C[r] + bia;
          h = h > 0.0f ? h : 0.0f;
          h1b[(r + g * 8) * 128 + nt * 16 + l16] = (_Float16)h;
        }
      }
    }
    // ================= forward L2: h1 @ W2 -> h2 ========================
    {
      v16h A[4];
#pragma unroll
      for (int kc = 0; kc < 4; ++kc) A[kc] = load_a16(h1b + l16 * 128, kc * 32, g);
#pragma unroll
      for (int nt = 0; nt < 8; ++nt) {
        const _Float16* brow = W2T + (nt * 16 + l16) * 128;
        v8f C = {};
#pragma unroll
        for (int kc = 0; kc < 4; ++kc)
          C = wmma_f16(A[kc], load_b16(brow, kc * 32, g), C);
        float bia = b2s[nt * 16 + l16];
#pragma unroll
        for (int r = 0; r < 8; ++r) {
          float h = C[r] + bia;
          h = h > 0.0f ? h : 0.0f;
          h2b[(r + g * 8) * 128 + nt * 16 + l16] = (_Float16)h;
        }
      }
    }
    // ============ forward L3: h2 @ W3 -> out[16x32]; v = vc + r =========
    float vt[8];
    {
      v8f O0 = {}, O1 = {};
#pragma unroll
      for (int kc = 0; kc < 4; ++kc) {
        v16h A = load_a16(h2b + l16 * 128, kc * 32, g);
        O0 = wmma_f16(A, load_b16(W3T + (l16)      * 128, kc * 32, g), O0);
        O1 = wmma_f16(A, load_b16(W3T + (16 + l16) * 128, kc * 32, g), O1);
      }
      float b3a = b3s[l16], b3b = b3s[16 + l16];
#pragma unroll
      for (int r = 0; r < 8; ++r) vt[r] = O0[r] + O1[r] + b3a + b3b;
    }

    // ========== backward: Rademacher probe u; g_out = [u, u] ============
    v16h uA;
    {
      const int* rp =
          radg + ((size_t)step * (size_t)Btot + (size_t)(rowbase + l16)) * 16 + g * 8;
      int4 ra = *(const int4*)(rp);
      int4 rb = *(const int4*)(rp + 4);
      v8h up;
      up[0] = (_Float16)(float)(2 * ra.x - 1);
      up[1] = (_Float16)(float)(2 * ra.y - 1);
      up[2] = (_Float16)(float)(2 * ra.z - 1);
      up[3] = (_Float16)(float)(2 * ra.w - 1);
      up[4] = (_Float16)(float)(2 * rb.x - 1);
      up[5] = (_Float16)(float)(2 * rb.y - 1);
      up[6] = (_Float16)(float)(2 * rb.z - 1);
      up[7] = (_Float16)(float)(2 * rb.w - 1);
#pragma unroll
      for (int i = 0; i < 8; ++i) { uA[i] = up[i]; uA[i + 8] = up[i]; }
      *(v8h*)(ub + l16 * 16 + g * 8) = up;  // C-layout readback for div
    }
    // ---- bwd L3: g_h2 = relu'(h2) * (g_out[16x32] @ W3^T) ----
#pragma unroll
    for (int nt = 0; nt < 8; ++nt) {
      v8f C = {};
      C = wmma_f16(uA, load_b16(W3R + (nt * 16 + l16) * 32, 0, g), C);
#pragma unroll
      for (int r = 0; r < 8; ++r) {
        _Float16 hv = h2b[(r + g * 8) * 128 + nt * 16 + l16];
        gb1[(r + g * 8) * 128 + nt * 16 + l16] =
            (hv > (_Float16)0.0f) ? (_Float16)C[r] : (_Float16)0.0f;
      }
    }
    // ---- bwd L2: g_h1 = relu'(h1) * (g_h2 @ W2^T) ----
    {
      v16h A[4];
#pragma unroll
      for (int kc = 0; kc < 4; ++kc) A[kc] = load_a16(gb1 + l16 * 128, kc * 32, g);
#pragma unroll
      for (int nt = 0; nt < 8; ++nt) {
        const _Float16* brow = W2R + (nt * 16 + l16) * 128;
        v8f C = {};
#pragma unroll
        for (int kc = 0; kc < 4; ++kc)
          C = wmma_f16(A[kc], load_b16(brow, kc * 32, g), C);
#pragma unroll
        for (int r = 0; r < 8; ++r) {
          _Float16 hv = h1b[(r + g * 8) * 128 + nt * 16 + l16];
          gb2[(r + g * 8) * 128 + nt * 16 + l16] =
              (hv > (_Float16)0.0f) ? (_Float16)C[r] : (_Float16)0.0f;
        }
      }
    }
    // ---- bwd L1: jv = g_h1 @ W1a^T  (only a-rows of W1) ----
    v8f J = {};
#pragma unroll
    for (int kc = 0; kc < 4; ++kc) {
      v16h A = load_a16(gb2 + l16 * 128, kc * 32, g);
      J = wmma_f16(A, load_b16(W1A + l16 * 128, kc * 32, g), J);
    }
    // div[M] = sum_j jv[M][j]*u[M][j]; Euler update
#pragma unroll
    for (int r = 0; r < 8; ++r) {
      float p = J[r] * (float)ub[(r + g * 8) * 16 + l16];
      p = red16(p);
      ld_reg[r] -= DELTA_F * p;
      a_reg[r]  += DELTA_F * vt[r];
    }
  }

  // logp = sum_j(-0.5 a^2) - 8*log(2pi) + log_det
#pragma unroll
  for (int r = 0; r < 8; ++r) {
    float q = red16(-0.5f * a_reg[r] * a_reg[r]);
    float res = q - 8.0f * LOG2PI_F + ld_reg[r];
    if (l16 == r) outg[rowbase + r + g * 8] = res;
  }
}

extern "C" void kernel_launch(void* const* d_in, const int* in_sizes, int n_in,
                              void* d_out, int out_size, void* d_ws, size_t ws_size,
                              hipStream_t stream) {
  (void)n_in; (void)out_size; (void)d_ws; (void)ws_size;
  const float* s   = (const float*)d_in[0];
  const float* a0  = (const float*)d_in[1];
  const int*   rad = (const int*)d_in[2];
  const float* W1  = (const float*)d_in[3];
  const float* b1  = (const float*)d_in[4];
  const float* W2  = (const float*)d_in[5];
  const float* b2  = (const float*)d_in[6];
  const float* W3  = (const float*)d_in[7];
  const float* b3  = (const float*)d_in[8];
  float* out = (float*)d_out;

  const int B = in_sizes[1] / 16;           // a0 is [B,16]
  const dim3 grid(B / 128), block(256);     // 8 waves x 16 rows per block
  const size_t shmem = 271488;              // 110.6KB weights + 8x19.9KB staging
  hipLaunchKernelGGL(cfm_loglik_kernel, grid, block, shmem, stream,
                     s, a0, rad, W1, b1, W2, b2, W3, b3, out, B);
}
